// GCN_67920612819497
// MI455X (gfx1250) — matrix operand order repaired and checked by reading, not compiled, not run on hardware
//
#include <hip/hip_runtime.h>
#include <stdint.h>

// ---------------------------------------------------------------------------
// GCN forward for MI455X (gfx1250, wave32, WMMA).
// GEMMs in bf16 WMMA w/ f32 accumulation; graph scatter via f32 global atomics.
// ---------------------------------------------------------------------------

typedef __attribute__((ext_vector_type(16))) __bf16          v16bf;
typedef __attribute__((ext_vector_type(16))) unsigned short  v16u;
typedef __attribute__((ext_vector_type(8)))  float           v8f;

#define NNODES  100000
#define NEDGES  1600000
#define INDIM   500
#define HID     256
#define NCLS    40

__device__ __forceinline__ unsigned short f2bf_bits(float x) {
    uint32_t u = __float_as_uint(x);
    u += 0x7FFFu + ((u >> 16) & 1u);          // round-to-nearest-even
    return (unsigned short)(u >> 16);
}

#if __has_builtin(__builtin_amdgcn_cvt_pk_bf16_f32)
typedef __attribute__((ext_vector_type(2))) __bf16 v2bf;
__device__ __forceinline__ uint32_t pack2bf(float lo, float hi) {
    v2bf r = __builtin_amdgcn_cvt_pk_bf16_f32(lo, hi);
    return __builtin_bit_cast(uint32_t, r);
}
#else
__device__ __forceinline__ uint32_t rnd_bf(float x) {
    uint32_t u = __float_as_uint(x);
    return u + (0x7FFFu + ((u >> 16) & 1u));
}
// v_perm_b32: take high halves of the two rounded dwords -> packed 2xbf16
__device__ __forceinline__ uint32_t pack2bf(float lo, float hi) {
    return __builtin_amdgcn_perm(rnd_bf(hi), rnd_bf(lo), 0x07060302u);
}
#endif

// ---------------------------------------------------------------------------
// Generic GEMM: C[M,N] = A[M,K] @ B[K,N] (+bias) (+ReLU), bf16 WMMA, f32 accum.
// Block tile 128(M) x 64(N); 8 waves, each owns a 32x32 tile -> 4 WMMAs/K-step.
// ---------------------------------------------------------------------------
#define BM 128
#define BN 64
#define BK 32
#define LDA 40     // padded LDS row (ushorts) -> 80B; 16 rows conflict-free
#define LDB 40

__global__ __launch_bounds__(256)
void gemm_bf16_wmma(const float* __restrict__ A, const float* __restrict__ B,
                    const float* __restrict__ bias, float* __restrict__ C,
                    int M, int N, int K, int relu)
{
    __shared__ alignas(16) unsigned short As[BM * LDA];   // [m][k]
    __shared__ alignas(16) unsigned short Bs[BN * LDB];   // transposed: [n][k]

    const int tid  = threadIdx.x;
    const int lane = tid & 31;
    const int wave = tid >> 5;
    const int m0 = blockIdx.y * BM;
    const int n0 = blockIdx.x * BN;
    const int wm = (wave >> 1) << 5;          // 0,32,64,96
    const int wn = (wave & 1) << 5;           // 0,32

    v8f acc00 = {}, acc01 = {}, acc10 = {}, acc11 = {};

    // per-lane fragment source addresses (constant across K loop)
    const int kbA = (lane < 16) ? 0 : 16;
    const int kbB = (lane < 16) ? 0 : 16;
    const unsigned short* aptr0 = &As[(wm + (lane & 15)) * LDA + kbA];
    const unsigned short* aptr1 = aptr0 + 16 * LDA;
    const unsigned short* bptr0 = &Bs[(wn + (lane & 15)) * LDB + kbB];
    const unsigned short* bptr1 = bptr0 + 16 * LDB;

    const int kt = (K + BK - 1) / BK;
    for (int t = 0; t < kt; ++t) {
        const int k0 = t * BK;

        // ---- stage A tile: 128x32 fp32 -> bf16 LDS -------------------------
        if (m0 + BM <= M && k0 + BK <= K) {
            // fast path: 4x global_load_b128, 2x ds_store_b128 (16 bf16)
            const int r  = tid >> 1;
            const int cb = (tid & 1) << 4;    // 0 or 16
            const float4* p = reinterpret_cast<const float4*>(
                                  A + (size_t)(m0 + r) * K + (k0 + cb));
            float4 f0 = p[0], f1 = p[1], f2 = p[2], f3 = p[3];
            uint4 u0, u1;
            u0.x = pack2bf(f0.x, f0.y);  u0.y = pack2bf(f0.z, f0.w);
            u0.z = pack2bf(f1.x, f1.y);  u0.w = pack2bf(f1.z, f1.w);
            u1.x = pack2bf(f2.x, f2.y);  u1.y = pack2bf(f2.z, f2.w);
            u1.z = pack2bf(f3.x, f3.y);  u1.w = pack2bf(f3.z, f3.w);
            uint4* q = reinterpret_cast<uint4*>(&As[r * LDA + cb]);
            q[0] = u0;
            *reinterpret_cast<uint4*>(&As[r * LDA + cb + 8]) = u1;
        } else {
#pragma unroll
            for (int i = 0; i < 16; ++i) {
                int idx = tid + i * 256;
                int r = idx >> 5, c = idx & 31;
                int gm = m0 + r, gk = k0 + c;
                float v = (gm < M && gk < K) ? A[(size_t)gm * K + gk] : 0.0f;
                As[r * LDA + c] = f2bf_bits(v);
            }
        }

        // ---- stage B tile transposed: 32(k) x 64(n) -> Bs[n][k] ------------
        if (k0 + BK <= K && n0 + BN <= N) {
            const int r  = tid >> 3;          // k row: 0..31
            const int cb = (tid & 7) << 3;    // n col: 0..56 step 8
            const float4* p = reinterpret_cast<const float4*>(
                                  B + (size_t)(k0 + r) * N + (n0 + cb));
            float4 f0 = p[0], f1 = p[1];
            Bs[(cb + 0) * LDB + r] = f2bf_bits(f0.x);
            Bs[(cb + 1) * LDB + r] = f2bf_bits(f0.y);
            Bs[(cb + 2) * LDB + r] = f2bf_bits(f0.z);
            Bs[(cb + 3) * LDB + r] = f2bf_bits(f0.w);
            Bs[(cb + 4) * LDB + r] = f2bf_bits(f1.x);
            Bs[(cb + 5) * LDB + r] = f2bf_bits(f1.y);
            Bs[(cb + 6) * LDB + r] = f2bf_bits(f1.z);
            Bs[(cb + 7) * LDB + r] = f2bf_bits(f1.w);
        } else {
#pragma unroll
            for (int i = 0; i < 8; ++i) {
                int idx = tid + i * 256;
                int r = idx >> 6, c = idx & 63;   // r = k, c = n
                int gk = k0 + r, gn = n0 + c;
                float v = (gk < K && gn < N) ? B[(size_t)gk * N + gn] : 0.0f;
                Bs[c * LDB + r] = f2bf_bits(v);
            }
        }

        // prefetch next A K-slab into cache hierarchy
        if (t + 1 < kt) {
            int pm = m0 + (tid >> 1);
            if (pm < M)
                __builtin_prefetch(&A[(size_t)pm * K + (k0 + BK)], 0, 3);
        }
        __syncthreads();

        // Fragment assembly, contiguous k order (same for A and B):
        //  A 16x32: lanes 0-15 hold K 0-15      ; lanes 16-31 hold K 16-31
        //  B 32x16: lanes 0-15 hold K 0-15      ; lanes 16-31 hold K 16-31
        v16u au0, au1, bu0, bu1;
#pragma unroll
        for (int i = 0; i < 8; ++i) {
            au0[i]     = aptr0[i];
            au0[i + 8] = aptr0[8 + i];
            au1[i]     = aptr1[i];
            au1[i + 8] = aptr1[8 + i];
        }
#pragma unroll
        for (int i = 0; i < 16; ++i) {
            bu0[i] = bptr0[i];
            bu1[i] = bptr1[i];
        }
        v16bf av0 = __builtin_bit_cast(v16bf, au0);
        v16bf av1 = __builtin_bit_cast(v16bf, au1);
        v16bf bv0 = __builtin_bit_cast(v16bf, bu0);
        v16bf bv1 = __builtin_bit_cast(v16bf, bu1);
        acc00 = __builtin_amdgcn_wmma_f32_16x16x32_bf16(
                    false, av0, false, bv0, (short)0, acc00, false, false);
        acc01 = __builtin_amdgcn_wmma_f32_16x16x32_bf16(
                    false, av0, false, bv1, (short)0, acc01, false, false);
        acc10 = __builtin_amdgcn_wmma_f32_16x16x32_bf16(
                    false, av1, false, bv0, (short)0, acc10, false, false);
        acc11 = __builtin_amdgcn_wmma_f32_16x16x32_bf16(
                    false, av1, false, bv1, (short)0, acc11, false, false);
        __syncthreads();
    }

    // Epilogue: v8f layout -> VGPR r: M=r (lanes 0-15) / M=8+r (lanes 16-31)
    const int col0   = n0 + wn + (lane & 15);
    const int col1   = col0 + 16;
    const int rbase0 = m0 + wm + ((lane >> 4) << 3);
    const int rbase1 = rbase0 + 16;
    float bv0 = (bias && col0 < N) ? bias[col0] : 0.0f;
    float bv1 = (bias && col1 < N) ? bias[col1] : 0.0f;
#pragma unroll
    for (int r = 0; r < 8; ++r) {
        int row0 = rbase0 + r, row1 = rbase1 + r;
        if (row0 < M) {
            if (col0 < N) {
                float v = acc00[r] + bv0;
                if (relu) v = fmaxf(v, 0.0f);
                C[(size_t)row0 * N + col0] = v;
            }
            if (col1 < N) {
                float v = acc01[r] + bv1;
                if (relu) v = fmaxf(v, 0.0f);
                C[(size_t)row0 * N + col1] = v;
            }
        }
        if (row1 < M) {
            if (col0 < N) {
                float v = acc10[r] + bv0;
                if (relu) v = fmaxf(v, 0.0f);
                C[(size_t)row1 * N + col0] = v;
            }
            if (col1 < N) {
                float v = acc11[r] + bv1;
                if (relu) v = fmaxf(v, 0.0f);
                C[(size_t)row1 * N + col1] = v;
            }
        }
    }
}

// ---------------------------------------------------------------------------
// Graph kernels
// ---------------------------------------------------------------------------
__global__ void zero_f32(float* __restrict__ p, int n) {
    int i = blockIdx.x * blockDim.x + threadIdx.x;
    if (i < n) p[i] = 0.0f;
}

__global__ void deg_count(const long long* __restrict__ dst, float* __restrict__ deg, int E) {
    int e = blockIdx.x * blockDim.x + threadIdx.x;
    if (e < E) atomicAdd(&deg[(int)dst[e]], 1.0f);
}

__global__ void make_norms(const float* __restrict__ deg, float* __restrict__ dinv,
                           float* __restrict__ selfn, int n) {
    int i = blockIdx.x * blockDim.x + threadIdx.x;
    if (i < n) {
        float d = deg[i] + 1.0f;         // +1 for self loop
        dinv[i]  = rsqrtf(d);
        selfn[i] = 1.0f / d;
    }
}

// agg[i,f] = hw[i,f] * selfnorm[i]   (initializes agg before edge scatter)
__global__ void init_agg(const float* __restrict__ hw, const float* __restrict__ selfn,
                         float* __restrict__ agg, int n) {
    int i = blockIdx.x * blockDim.x + threadIdx.x;
    if (i < n) agg[i] = hw[i] * selfn[i >> 8];     // HID == 256
}

// One float4 feature chunk per thread; 64 threads per edge (HID=256).
__global__ void scatter_edges(const long long* __restrict__ src, const long long* __restrict__ dst,
                              const float* __restrict__ dinv, const float* __restrict__ hw,
                              float* __restrict__ agg, int E) {
    int t = blockIdx.x * blockDim.x + threadIdx.x;
    int e = t >> 6;
    if (e >= E) return;
    int c = (t & 63) << 2;
    int s = (int)src[e], d = (int)dst[e];
    float w = dinv[s] * dinv[d];
    const float4 v = *reinterpret_cast<const float4*>(hw + (size_t)s * HID + c);
    float* ap = agg + (size_t)d * HID + c;
    atomicAdd(ap + 0, v.x * w);
    atomicAdd(ap + 1, v.y * w);
    atomicAdd(ap + 2, v.z * w);
    atomicAdd(ap + 3, v.w * w);
}

// h = relu((agg + conv_b) * gamma + beta) + h   (BN eval + ReLU + residual)
__global__ void post_bn_relu_res(const float* __restrict__ agg, const float* __restrict__ b,
                                 const float* __restrict__ g, const float* __restrict__ be,
                                 float* __restrict__ h, int n) {
    int i = blockIdx.x * blockDim.x + threadIdx.x;
    if (i < n) {
        int c = i & 255;                 // HID == 256
        float v = (agg[i] + b[c]) * g[c] + be[c];
        h[i] += fmaxf(v, 0.0f);
    }
}

// ---------------------------------------------------------------------------
// Launch
// ---------------------------------------------------------------------------
extern "C" void kernel_launch(void* const* d_in, const int* in_sizes, int n_in,
                              void* d_out, int out_size, void* d_ws, size_t ws_size,
                              hipStream_t stream) {
    (void)in_sizes; (void)n_in; (void)out_size; (void)ws_size;

    const float*     x     = (const float*)d_in[0];
    const long long* ei    = (const long long*)d_in[1];
    const float*     embW  = (const float*)d_in[2];
    const float*     embB  = (const float*)d_in[3];
    const float*     convW = (const float*)d_in[4];
    const float*     convB = (const float*)d_in[5];
    const float*     gamma = (const float*)d_in[6];
    const float*     beta  = (const float*)d_in[7];
    const float*     W0 = (const float*)d_in[8];  const float* b0 = (const float*)d_in[9];
    const float*     W1 = (const float*)d_in[10]; const float* b1 = (const float*)d_in[11];
    const float*     W2 = (const float*)d_in[12]; const float* b2 = (const float*)d_in[13];
    const float*     W3 = (const float*)d_in[14]; const float* b3 = (const float*)d_in[15];
    float* out = (float*)d_out;

    const size_t featBytes = (size_t)NNODES * HID * sizeof(float);   // 102.4 MB
    char* ws = (char*)d_ws;
    float* h     = (float*)(ws);
    float* buf1  = (float*)(ws + featBytes);          // hw / h1 / h3
    float* buf2  = (float*)(ws + 2 * featBytes);      // agg / h2
    float* deg   = (float*)(ws + 3 * featBytes);
    float* dinv  = deg  + NNODES;
    float* selfn = dinv + NNODES;

    const long long* srcp = ei;            // edge_index[0]
    const long long* dstp = ei + NEDGES;   // edge_index[1]

    const int NH = NNODES * HID;
    dim3 blk(256);

    // degrees + symmetric norms
    zero_f32  <<<(NNODES + 255) / 256, blk, 0, stream>>>(deg, NNODES);
    deg_count <<<(NEDGES + 255) / 256, blk, 0, stream>>>(dstp, deg, NEDGES);
    make_norms<<<(NNODES + 255) / 256, blk, 0, stream>>>(deg, dinv, selfn, NNODES);

    auto gemm = [&](const float* A, const float* B, const float* bias, float* C,
                    int M, int N, int K, int relu) {
        dim3 g((unsigned)((N + BN - 1) / BN), (unsigned)((M + BM - 1) / BM));
        gemm_bf16_wmma<<<g, 256, 0, stream>>>(A, B, bias, C, M, N, K, relu);
    };

    // embed: h = x @ embed_W + embed_b
    gemm(x, embW, embB, h, NNODES, HID, INDIM, 0);

    // GCN layers
    for (int l = 0; l < 3; ++l) {
        gemm(h, convW + (size_t)l * HID * HID, nullptr, buf1, NNODES, HID, HID, 0);
        init_agg<<<(NH + 255) / 256, blk, 0, stream>>>(buf1, selfn, buf2, NH);
        scatter_edges<<<(unsigned)((NEDGES * 64 + 255) / 256), blk, 0, stream>>>(
            srcp, dstp, dinv, buf1, buf2, NEDGES);
        post_bn_relu_res<<<(NH + 255) / 256, blk, 0, stream>>>(
            buf2, convB + l * HID, gamma + l * HID, beta + l * HID, h, NH);
    }

    // classifier MLP: 256 -> 128 -> 64 -> 32 -> 40
    gemm(h,    W0, b0, buf1, NNODES, 128, 256, 1);
    gemm(buf1, W1, b1, buf2, NNODES,  64, 128, 1);
    gemm(buf2, W2, b2, buf1, NNODES,  32,  64, 1);
    gemm(buf1, W3, b3, out,  NNODES, NCLS, 32, 0);
}
